// pointnet2_cls_80083960201386
// MI455X (gfx1250) — compile-verified, matrix operand order checked by reading, not run on hardware
//
#include <hip/hip_runtime.h>
#include <hip/hip_bf16.h>

typedef _Float16 half_t;
typedef __attribute__((ext_vector_type(16))) _Float16 v16h;
typedef __attribute__((ext_vector_type(8)))  _Float16 v8h;
typedef __attribute__((ext_vector_type(8)))  float    v8f;

// ---------------------------------------------------------------------------
// kNN: one thread per query point, brute force over Npts candidates,
// "replace current worst" selection. Index order is irrelevant (max-pool).
// ---------------------------------------------------------------------------
template <int K>
__global__ void knn_kernel(const float* __restrict__ xyz, int Npts, int S,
                           int* __restrict__ idx) {
    int t = blockIdx.x * blockDim.x + threadIdx.x;
    int total = 16 * S;
    if (t >= total) return;
    int b = t / S, s = t % S;
    const float* base = xyz + (size_t)b * 4096 * 3;
    float cx = base[s * 3 + 0], cy = base[s * 3 + 1], cz = base[s * 3 + 2];
    float bd[K]; int bi[K];
#pragma unroll
    for (int j = 0; j < K; ++j) { bd[j] = 3.0e38f; bi[j] = 0; }
    float worst = 3.0e38f; int wpos = 0;
    for (int n = 0; n < Npts; ++n) {
        float dx = base[n * 3 + 0] - cx;
        float dy = base[n * 3 + 1] - cy;
        float dz = base[n * 3 + 2] - cz;
        float d = dx * dx + dy * dy + dz * dz;
        if (d < worst) {
            bd[wpos] = d; bi[wpos] = n;
            worst = bd[0]; wpos = 0;
#pragma unroll
            for (int j = 1; j < K; ++j)
                if (bd[j] > worst) { worst = bd[j]; wpos = j; }
        }
    }
    int* o = idx + (size_t)t * K;
#pragma unroll
    for (int j = 0; j < K; ++j) o[j] = bi[j];
}

// ---------------------------------------------------------------------------
// Grouping kernels. Layouts (must match pack_weight_kernel's row remap):
//   SA1: [rel_xyz(3) | pts_xyz(3) | 0.. to 32]                 (no shift)
//   SA2: [rel_xyz(3) | 0 | l1_feat(128) | 0.. to 160]          (shift)
//   SA3: [xyz(3)     | 0 | l2_feat(256) | 0.. to 288]          (shift)
// The pad-after-xyz aligns the feature block to 8B for uint2 (b64) copies.
// ---------------------------------------------------------------------------
__global__ void group1_kernel(const float* __restrict__ pts,
                              const int* __restrict__ idx,
                              half_t* __restrict__ X) {
    int r = blockIdx.x * blockDim.x + threadIdx.x;
    if (r >= 16 * 512 * 32) return;
    int s = (r >> 5) & 511;
    int b = r >> 14;
    const float* base = pts + (size_t)b * 4096 * 3;
    int i = idx[r];
    float cx = base[s * 3 + 0], cy = base[s * 3 + 1], cz = base[s * 3 + 2];
    float px = base[i * 3 + 0], py = base[i * 3 + 1], pz = base[i * 3 + 2];
    half_t* row = X + (size_t)r * 32;
    row[0] = (half_t)(px - cx); row[1] = (half_t)(py - cy); row[2] = (half_t)(pz - cz);
    row[3] = (half_t)px;        row[4] = (half_t)py;        row[5] = (half_t)pz;
    row[6] = (half_t)0.f;       row[7] = (half_t)0.f;
    uint2 z = {0u, 0u};
    uint2* zp = (uint2*)(void*)(row + 8);        // 16B offset, 8B aligned
#pragma unroll
    for (int c = 0; c < 6; ++c) zp[c] = z;       // cols 8..31
}

__global__ void group2_kernel(const float* __restrict__ pts,
                              const int* __restrict__ idx,
                              const half_t* __restrict__ l1_feat,
                              half_t* __restrict__ X) {
    int r = blockIdx.x * blockDim.x + threadIdx.x;
    if (r >= 16 * 128 * 64) return;
    int s = (r >> 6) & 127;
    int b = r >> 13;
    const float* base = pts + (size_t)b * 4096 * 3;
    int i = idx[r];  // in [0, 512)
    float cx = base[s * 3 + 0], cy = base[s * 3 + 1], cz = base[s * 3 + 2];
    float px = base[i * 3 + 0], py = base[i * 3 + 1], pz = base[i * 3 + 2];
    half_t* row = X + (size_t)r * 160;
    row[0] = (half_t)(px - cx); row[1] = (half_t)(py - cy); row[2] = (half_t)(pz - cz);
    row[3] = (half_t)0.f;
    const uint2* src = (const uint2*)(void*)(l1_feat + ((size_t)b * 512 + i) * 128);
    uint2* dst = (uint2*)(void*)(row + 4);
#pragma unroll 8
    for (int c = 0; c < 32; ++c) dst[c] = src[c];   // 128 halves
    uint2 z = {0u, 0u};
    uint2* zp = (uint2*)(void*)(row + 132);
#pragma unroll
    for (int c = 0; c < 7; ++c) zp[c] = z;          // cols 132..159
}

__global__ void group3_kernel(const float* __restrict__ pts,
                              const half_t* __restrict__ l2_feat,
                              half_t* __restrict__ X) {
    int r = blockIdx.x * blockDim.x + threadIdx.x;
    if (r >= 16 * 128) return;
    int j = r & 127;
    int b = r >> 7;
    const float* base = pts + (size_t)b * 4096 * 3;
    half_t* row = X + (size_t)r * 288;
    row[0] = (half_t)base[j * 3 + 0];
    row[1] = (half_t)base[j * 3 + 1];
    row[2] = (half_t)base[j * 3 + 2];
    row[3] = (half_t)0.f;
    const uint2* src = (const uint2*)(void*)(l2_feat + (size_t)r * 256);
    uint2* dst = (uint2*)(void*)(row + 4);
#pragma unroll 8
    for (int c = 0; c < 64; ++c) dst[c] = src[c];   // 256 halves
    uint2 z = {0u, 0u};
    uint2* zp = (uint2*)(void*)(row + 260);
#pragma unroll
    for (int c = 0; c < 7; ++c) zp[c] = z;          // cols 260..287
}

// ---------------------------------------------------------------------------
// Weight prep: f32 (cin x cout) -> f16 packed into WMMA B-fragment order.
// Tile (kt, nt) is a 32x16 KxN block; element (lane, j) holds
//   k = kt*32 + (j&7) + ((j>>3)<<4) + 8*(lane>=16),  n = nt*16 + (lane&15)
// stored at Wp[((nt*Kt + kt)*32 + lane)*16 + j], zero-padded in K and N.
// shift=1 remaps padded input rows: k<3 -> k, k==3 -> zero, k>3 -> k-1.
// ---------------------------------------------------------------------------
__global__ void pack_weight_kernel(const float* __restrict__ W,
                                   half_t* __restrict__ Wp,
                                   int cin, int cout, int Kt, int Nt, int shift) {
    int t = blockIdx.x * blockDim.x + threadIdx.x;
    if (t >= Kt * Nt * 512) return;
    int j    = t & 15;
    int lane = (t >> 4) & 31;
    int tile = t >> 9;                 // nt*Kt + kt
    int kt = tile % Kt, nt = tile / Kt;
    int k = kt * 32 + (j & 7) + ((j >> 3) << 4) + ((lane >> 4) << 3);
    int n = nt * 16 + (lane & 15);
    int ks = k;
    if (shift) { if (k == 3) ks = -1; else if (k > 3) ks = k - 1; }
    half_t v = (half_t)0.f;
    if (ks >= 0 && ks < cin && n < cout) v = (half_t)W[(size_t)ks * cout + n];
    Wp[t] = v;
}

// ---------------------------------------------------------------------------
// WMMA GEMM, 16x64 output per wave (4 N-tiles share one A fragment):
//   Y = relu(g * (X@W + b) + bt)   (gamma != nullptr, f16 output)
//   Y = X@W + b                    (gamma == nullptr, f32 output)
// A fragment: two contiguous b128 loads per lane (layout per ISA 7.12.2).
// B fragment: two contiguous b128 loads from the packed weight buffer.
// M must be a multiple of 16 (true at all call sites); N may be ragged.
// ---------------------------------------------------------------------------
__global__ __launch_bounds__(32)
void wmma_gemm_kernel(const half_t* __restrict__ X, const half_t* __restrict__ Wp,
                      const float* __restrict__ bias,
                      const float* __restrict__ gamma,
                      const float* __restrict__ beta,
                      half_t* __restrict__ Yh, float* __restrict__ Yf,
                      int M, int N, int Kpad) {
    const int lane = threadIdx.x & 31;
    const int hi   = lane >> 4;           // 0 or 1
    const int lm   = lane & 15;
    const int rowA = blockIdx.x * 16 + lm;
    const int Kt      = Kpad >> 5;
    const int Ntiles  = (N + 15) >> 4;
    const int nt0     = blockIdx.y * 4;

    const half_t* xrow = X + (size_t)rowA * Kpad + (hi << 3);

    v8f acc0 = {}, acc1 = {}, acc2 = {}, acc3 = {};
    for (int kt = 0; kt < Kt; ++kt) {
        if (kt + 1 < Kt)   // gfx1250 global_prefetch_b8 on the next A slab
            __builtin_prefetch(xrow + (kt + 1) * 32, 0, 1);
        v8h alo = *(const v8h*)(xrow + kt * 32);
        v8h ahi = *(const v8h*)(xrow + kt * 32 + 16);
        v16h a = __builtin_shufflevector(alo, ahi,
                 0, 1, 2, 3, 4, 5, 6, 7, 8, 9, 10, 11, 12, 13, 14, 15);
#pragma unroll
        for (int t = 0; t < 4; ++t) {
            int nt = nt0 + t;
            if (nt >= Ntiles) nt = Ntiles - 1;   // clamp; store is masked later
            const half_t* bp = Wp + (((size_t)nt * Kt + kt) * 32 + lane) * 16;
            v8h blo = *(const v8h*)bp;
            v8h bhi = *(const v8h*)(bp + 8);
            v16h bv = __builtin_shufflevector(blo, bhi,
                      0, 1, 2, 3, 4, 5, 6, 7, 8, 9, 10, 11, 12, 13, 14, 15);
            v8f c = (t == 0) ? acc0 : (t == 1) ? acc1 : (t == 2) ? acc2 : acc3;
            c = __builtin_amdgcn_wmma_f32_16x16x32_f16(
                false, a, false, bv, (short)0, c, false, false);
            if (t == 0) acc0 = c; else if (t == 1) acc1 = c;
            else if (t == 2) acc2 = c; else acc3 = c;
        }
    }

#pragma unroll
    for (int t = 0; t < 4; ++t) {
        int nt = nt0 + t;
        if (nt >= Ntiles) break;
        int col = nt * 16 + lm;
        if (col >= N) continue;
        v8f c = (t == 0) ? acc0 : (t == 1) ? acc1 : (t == 2) ? acc2 : acc3;
        const float b  = bias[col];
        const float g  = gamma ? gamma[col] : 0.f;
        const float bt = beta  ? beta[col]  : 0.f;
#pragma unroll
        for (int r = 0; r < 8; ++r) {
            int row = blockIdx.x * 16 + r + (hi << 3);
            float v = c[r] + b;
            if (gamma) {
                v = g * v + bt;
                v = v > 0.f ? v : 0.f;
                Yh[(size_t)row * N + col] = (half_t)v;
            } else {
                Yf[(size_t)row * N + col] = v;
            }
        }
    }
}

// ---------------------------------------------------------------------------
// Max-pool over the K grouped rows of each group, per channel.
// ---------------------------------------------------------------------------
__global__ void maxpool_kernel(const half_t* __restrict__ Y,
                               half_t* __restrict__ out,
                               int groups, int Kg, int C) {
    int t = blockIdx.x * blockDim.x + threadIdx.x;
    if (t >= groups * C) return;
    int g = t / C, c = t % C;
    const half_t* p = Y + (size_t)g * Kg * C + c;
    float m = -3.0e38f;
    for (int k = 0; k < Kg; ++k) {
        float v = (float)p[(size_t)k * C];
        m = v > m ? v : m;
    }
    out[t] = (half_t)m;
}

// ---------------------------------------------------------------------------
// log_softmax over 16 rows x 40 classes -> d_out (f32)
// ---------------------------------------------------------------------------
__global__ void log_softmax_kernel(const float* __restrict__ scores,
                                   float* __restrict__ out) {
    int row = threadIdx.x;
    if (row >= 16) return;
    const float* s = scores + row * 40;
    float m = -3.0e38f;
    for (int c = 0; c < 40; ++c) m = s[c] > m ? s[c] : m;
    float sum = 0.f;
    for (int c = 0; c < 40; ++c) sum += __expf(s[c] - m);
    float lse = m + __logf(sum);
    for (int c = 0; c < 40; ++c) out[row * 40 + c] = s[c] - lse;
}

// ---------------------------------------------------------------------------
extern "C" void kernel_launch(void* const* d_in, const int* in_sizes, int n_in,
                              void* d_out, int out_size, void* d_ws, size_t ws_size,
                              hipStream_t stream) {
    (void)in_sizes; (void)n_in; (void)out_size; (void)ws_size;

    const float* pts = (const float*)d_in[0];

    // ---- workspace carve-out ------------------------------------------------
    char* ws = (char*)d_ws;
    size_t off = 0;
    auto take = [&](size_t bytes) -> void* {
        void* p = ws + off;
        off = (off + bytes + 255) & ~(size_t)255;
        return p;
    };
    const size_t ROWS1 = 16 * 512 * 32;   // 262144
    const size_t ROWS2 = 16 * 128 * 64;   // 131072
    const size_t ROWS3 = 16 * 128;        // 2048

    int*    idx1   = (int*)take(ROWS1 * sizeof(int));
    int*    idx2   = (int*)take(ROWS2 * sizeof(int));
    half_t* bufA   = (half_t*)take(ROWS2 * 160 * sizeof(half_t)); // 40 MB peak
    half_t* bufB   = (half_t*)take(ROWS1 * 128 * sizeof(half_t)); // 64 MB peak
    half_t* l1f    = (half_t*)take((size_t)16 * 512 * 128 * sizeof(half_t));
    half_t* l2f    = (half_t*)take((size_t)16 * 128 * 256 * sizeof(half_t));
    half_t* gfeat  = (half_t*)take((size_t)16 * 1024 * sizeof(half_t));
    half_t* h1     = (half_t*)take((size_t)16 * 512 * sizeof(half_t));
    half_t* h2     = (half_t*)take((size_t)16 * 256 * sizeof(half_t));
    float*  scores = (float*)take((size_t)16 * 40 * sizeof(float));

    // ---- weight packing (f32 -> fragment-ordered, padded f16) ---------------
    struct WDesc { int in_idx, cin, cout, cinpad, shift; };
    const WDesc wd[12] = {
        {1,    6,   64,   32, 0}, {5,   64,   64,   64, 0}, {9,   64,  128,   64, 0},
        {13, 131,  128,  160, 1}, {17, 128,  128,  128, 0}, {21, 128,  256,  128, 0},
        {25, 259,  256,  288, 1}, {29, 256,  512,  256, 0}, {33, 512, 1024,  512, 0},
        {37,1024,  512, 1024, 0}, {41, 512,  256,  512, 0}, {45, 256,   40,  256, 0},
    };
    half_t* wptr[12];
    for (int i = 0; i < 12; ++i) {
        int Kt = wd[i].cinpad >> 5;
        int Nt = (wd[i].cout + 15) >> 4;
        size_t elems = (size_t)Kt * Nt * 512;
        wptr[i] = (half_t*)take(elems * sizeof(half_t));
        int n = (int)elems;
        pack_weight_kernel<<<(n + 255) / 256, 256, 0, stream>>>(
            (const float*)d_in[wd[i].in_idx], wptr[i],
            wd[i].cin, wd[i].cout, Kt, Nt, wd[i].shift);
    }

    auto gemm = [&](const half_t* X, int widx, half_t* Yh, float* Yf,
                    int M, int N, int Kpad, bool bn) {
        const float* b  = (const float*)d_in[wd[widx].in_idx + 1];
        const float* g  = bn ? (const float*)d_in[wd[widx].in_idx + 2] : nullptr;
        const float* bt = bn ? (const float*)d_in[wd[widx].in_idx + 3] : nullptr;
        int Ntiles = (N + 15) >> 4;
        dim3 grid(M / 16, (Ntiles + 3) / 4);
        wmma_gemm_kernel<<<grid, 32, 0, stream>>>(X, wptr[widx], b, g, bt,
                                                  Yh, Yf, M, N, Kpad);
    };

    // ---- SA1 ----------------------------------------------------------------
    knn_kernel<32><<<(16 * 512 + 255) / 256, 256, 0, stream>>>(pts, 4096, 512, idx1);
    group1_kernel<<<((int)ROWS1 + 255) / 256, 256, 0, stream>>>(pts, idx1, bufA);
    gemm(bufA, 0, bufB, nullptr, (int)ROWS1,  64,  32, true);
    gemm(bufB, 1, bufA, nullptr, (int)ROWS1,  64,  64, true);
    gemm(bufA, 2, bufB, nullptr, (int)ROWS1, 128,  64, true);
    maxpool_kernel<<<(16 * 512 * 128 + 255) / 256, 256, 0, stream>>>(
        bufB, l1f, 16 * 512, 32, 128);

    // ---- SA2 ----------------------------------------------------------------
    knn_kernel<64><<<(16 * 128 + 255) / 256, 256, 0, stream>>>(pts, 512, 128, idx2);
    group2_kernel<<<((int)ROWS2 + 255) / 256, 256, 0, stream>>>(pts, idx2, l1f, bufA);
    gemm(bufA, 3, bufB, nullptr, (int)ROWS2, 128, 160, true);
    gemm(bufB, 4, bufA, nullptr, (int)ROWS2, 128, 128, true);
    gemm(bufA, 5, bufB, nullptr, (int)ROWS2, 256, 128, true);
    maxpool_kernel<<<(16 * 128 * 256 + 255) / 256, 256, 0, stream>>>(
        bufB, l2f, 16 * 128, 64, 256);

    // ---- SA3 ----------------------------------------------------------------
    group3_kernel<<<((int)ROWS3 + 255) / 256, 256, 0, stream>>>(pts, l2f, bufA);
    gemm(bufA, 6, bufB, nullptr, (int)ROWS3,  256, 288, true);
    gemm(bufB, 7, bufA, nullptr, (int)ROWS3,  512, 256, true);
    gemm(bufA, 8, bufB, nullptr, (int)ROWS3, 1024, 512, true);
    maxpool_kernel<<<(16 * 1024 + 255) / 256, 256, 0, stream>>>(
        bufB, gfeat, 16, 128, 1024);

    // ---- FC head ------------------------------------------------------------
    gemm(gfeat,  9, h1, nullptr, 16, 512, 1024, true);
    gemm(h1,    10, h2, nullptr, 16, 256,  512, true);
    gemm(h2,    11, nullptr, scores, 16, 40, 256, false);  // linear, f32 out

    log_softmax_kernel<<<1, 32, 0, stream>>>(scores, (float*)d_out);
}
